// Model_65618510348706
// MI455X (gfx1250) — matrix operand order repaired and checked by reading, not compiled, not run on hardware
//
#include <hip/hip_runtime.h>

// ---------------------------------------------------------------------------
// Strided attention for MI455X/gfx1250, two-kernel pipeline:
//   1) prep: fp32 -> bf16 convert + sequence permutation; K row-major,
//      V transposed to [E][T] so the main kernel's B-fragments are contiguous.
//   2) attn: flash-attention v2 with v_wmma_f32_16x16x32_bf16; K/V^T tiles are
//      staged LDS-side by the Tensor Data Mover (tensor_load_to_lds), double
//      buffered, synchronized with s_wait_tensorcnt + workgroup barriers.
// Workgroup: 256 threads (8 waves) per 64-query tile.
//   wave w: row-tile = w/2 (16 queries), d-half = w%2 (256 of 512 out columns)
// ---------------------------------------------------------------------------

typedef __attribute__((ext_vector_type(16))) __bf16 v16bf;
typedef __attribute__((ext_vector_type(8)))  __bf16 v8bf;
typedef __attribute__((ext_vector_type(4)))  __bf16 v4bf;
typedef __attribute__((ext_vector_type(8)))  float  v8f;
typedef __attribute__((ext_vector_type(4)))  unsigned int u32x4;
typedef __attribute__((ext_vector_type(8)))  int      i32x8;
typedef __attribute__((ext_vector_type(4)))  int      i32x4;

#define T_CTX 4096
#define EMBD  512
#define NK    32      // kv tokens per pipeline step
#define MB    64      // query rows per workgroup

__device__ __forceinline__ int sperm(int t) {
    // (t % 64) * 64 + (t / 64), involution on [0,4096)
    return ((t & 63) << 6) | (t >> 6);
}

__device__ __forceinline__ v16bf ld_frag(const __bf16* p0, const __bf16* p1) {
    union { v16bf f; v8bf h[2]; } u;
    u.h[0] = *(const v8bf*)p0;   // ds_load_b128
    u.h[1] = *(const v8bf*)p1;   // ds_load_b128
    return u.f;
}

__device__ __forceinline__ float redmax16(float x) {
    x = fmaxf(x, __shfl_xor(x, 1, 32));
    x = fmaxf(x, __shfl_xor(x, 2, 32));
    x = fmaxf(x, __shfl_xor(x, 4, 32));
    x = fmaxf(x, __shfl_xor(x, 8, 32));
    return x;
}
__device__ __forceinline__ float redsum16(float x) {
    x += __shfl_xor(x, 1, 32);
    x += __shfl_xor(x, 2, 32);
    x += __shfl_xor(x, 4, 32);
    x += __shfl_xor(x, 8, 32);
    return x;
}

// ---- Tensor Data Mover: 2-D tile (bf16) global -> LDS -------------------
// D# group0: [1:0]=count=1, [63:32]=lds byte addr, [120:64]=global byte addr,
//            [127:126]=type 2. group1: [17:16]=data_size(1=2B), [79:48]=dim0,
//            [111:80]=dim1, [127:112]=tile_dim0, [143:128]=tile_dim1,
//            [207:160]=dim0_stride. 2-D only -> groups 2/3 zero.
// This toolchain's builtin takes 6 args (extra i32x8 group before cpol).
__device__ __forceinline__ void tdm_load_2d(const __bf16* gptr, unsigned lds_off,
                                            unsigned dim0, unsigned dim1,
                                            unsigned tile0, unsigned tile1,
                                            unsigned stride0) {
    unsigned long long ga = (unsigned long long)(size_t)gptr;
    u32x4 g0;
    g0[0] = 1u;                                         // count=1, user mode
    g0[1] = lds_off;                                    // LDS byte address
    g0[2] = (unsigned)ga;                               // global addr [31:0]
    g0[3] = (unsigned)((ga >> 32) & 0x1FFFFFFu) | (2u << 30);  // addr hi | type=2
    i32x8 g1;
    g1[0] = (int)(1u << 16);                            // data_size = 2 bytes
    g1[1] = (int)((dim0 & 0xFFFFu) << 16);              // tensor_dim0 lo
    g1[2] = (int)((dim0 >> 16) | ((dim1 & 0xFFFFu) << 16));
    g1[3] = (int)((dim1 >> 16) | (tile0 << 16));        // tile_dim0
    g1[4] = (int)tile1;                                 // tile_dim1 (tile_dim2=0)
    g1[5] = (int)stride0;                               // tensor_dim0_stride lo
    g1[6] = 0;
    g1[7] = 0;
    i32x4 z4 = {0, 0, 0, 0};
    i32x8 z8 = {0, 0, 0, 0, 0, 0, 0, 0};
    __builtin_amdgcn_tensor_load_to_lds(g0, g1, z4, z4, z8, 0);
}

// ---------------------------------------------------------------------------
// Kernel 1: permute + fp32->bf16.  Qp/Kp: [4][T][E] row-major (rows gathered
// through perm).  Vtp: [4][E][T] (transposed, rows gathered through perm).
// Grid: 4 batches x 64 token tiles; 256 threads.
// ---------------------------------------------------------------------------
__global__ __launch_bounds__(256, 2)
void prep_kernel(const float* __restrict__ Q, const float* __restrict__ K,
                 const float* __restrict__ V,
                 unsigned short* __restrict__ Qp_, unsigned short* __restrict__ Kp_,
                 unsigned short* __restrict__ Vtp_) {
    __bf16* Qp  = (__bf16*)Qp_;
    __bf16* Kp  = (__bf16*)Kp_;
    __bf16* Vtp = (__bf16*)Vtp_;
    __shared__ __bf16 tl[64][72];           // 64x64 transpose tile (+pad)

    const int tid = threadIdx.x;
    const int b   = blockIdx.x >> 6;
    const int t0  = (blockIdx.x & 63) << 6;
    const size_t bb = (size_t)b * T_CTX * EMBD;

    const int row = tid >> 2;               // 0..63 token-in-tile
    const int l4  = tid & 3;
    const size_t srow = bb + (size_t)sperm(t0 + row) * EMBD;

    // Q and K: straight permuted rows, coalesced float4 -> v4bf
    {
        const float4* qs = (const float4*)(Q + srow);
        const float4* ks = (const float4*)(K + srow);
        v4bf* qd = (v4bf*)(Qp + bb + (size_t)(t0 + row) * EMBD);
        v4bf* kd = (v4bf*)(Kp + bb + (size_t)(t0 + row) * EMBD);
        #pragma unroll
        for (int j = 0; j < 32; ++j) {
            const int c4 = j * 4 + l4;
            float4 f = qs[c4];
            qd[c4] = (v4bf){(__bf16)f.x, (__bf16)f.y, (__bf16)f.z, (__bf16)f.w};
            float4 g = ks[c4];
            kd[c4] = (v4bf){(__bf16)g.x, (__bf16)g.y, (__bf16)g.z, (__bf16)g.w};
        }
    }

    // V: 64x64 tiles through LDS -> Vtp[b][e][t]
    const float4* vsrc = (const float4*)(V + srow);
    for (int ec = 0; ec < 8; ++ec) {
        __syncthreads();
        #pragma unroll
        for (int j = 0; j < 4; ++j) {
            const int c4 = j * 4 + l4;                  // 0..15 within 64 cols
            float4 f = vsrc[ec * 16 + c4];
            *(v4bf*)(&tl[row][c4 * 4]) =
                (v4bf){(__bf16)f.x, (__bf16)f.y, (__bf16)f.z, (__bf16)f.w};
        }
        __syncthreads();
        const int el = tid >> 2;                        // 0..63 e-in-chunk
        #pragma unroll
        for (int j = 0; j < 4; ++j) {
            const int tk = (j * 4 + l4) * 4;            // token group of 4
            v4bf o = {tl[tk + 0][el], tl[tk + 1][el], tl[tk + 2][el], tl[tk + 3][el]};
            *(v4bf*)(Vtp + bb + (size_t)(ec * 64 + el) * T_CTX + t0 + tk) = o;
        }
    }
}

// ---------------------------------------------------------------------------
// Kernel 2: flash attention on preconverted bf16 operands, TDM-staged tiles.
// ---------------------------------------------------------------------------
__global__ __launch_bounds__(256, 1)
void strided_attn_fa(const unsigned short* __restrict__ Qp_,
                     const unsigned short* __restrict__ Kp_,
                     const unsigned short* __restrict__ Vtp_,
                     float* __restrict__ Out) {
    const __bf16* Qp  = (const __bf16*)Qp_;
    const __bf16* Kp  = (const __bf16*)Kp_;
    const __bf16* Vtp = (const __bf16*)Vtp_;

    extern __shared__ __align__(16) char smem[];
    __bf16* Qs = (__bf16*)smem;              // [64][512]        64 KB
    __bf16* Kb = Qs + MB * EMBD;             // 2 x [32][512]    64 KB
    __bf16* Vb = Kb + 2 * NK * EMBD;         // 2 x [512][32]    64 KB
    __bf16* Ps = Vb + 2 * EMBD * NK;         // [8][16][32]       8 KB

    const int tid   = threadIdx.x;
    const int lane  = tid & 31;
    const int w     = tid >> 5;
    const int g     = lane >> 4;
    const int ln    = lane & 15;
    const int rtile = w >> 1;
    const int dbase = (w & 1) * 256;

    const int b  = blockIdx.x >> 6;
    const int m0 = (blockIdx.x & 63) * MB;
    const size_t bb = (size_t)b * T_CTX * EMBD;

    const unsigned qs_off = (unsigned)(size_t)(void*)Qs;
    const unsigned kb_off = (unsigned)(size_t)(void*)Kb;
    const unsigned vb_off = (unsigned)(size_t)(void*)Vb;

    // ---- prologue: TDM loads for Q tile and KV tile 0 (wave 0 issues) ----
    if (w == 0) {
        tdm_load_2d(Qp + bb + (size_t)m0 * EMBD, qs_off, EMBD, T_CTX, EMBD, MB, EMBD);
        tdm_load_2d(Kp + bb, kb_off, EMBD, T_CTX, EMBD, NK, EMBD);
        tdm_load_2d(Vtp + bb, vb_off, T_CTX, EMBD, NK, EMBD, T_CTX);
    }

    const v8f zero = {0.f, 0.f, 0.f, 0.f, 0.f, 0.f, 0.f, 0.f};
    v8f acc[16];
    #pragma unroll
    for (int i = 0; i < 16; ++i) acc[i] = zero;
    float mrow[8], lrow[8];
    #pragma unroll
    for (int r = 0; r < 8; ++r) { mrow[r] = -1e30f; lrow[r] = 0.f; }

    const float C2 = 0.125f * 1.44269504088896f;   // scale * log2(e)
    const __bf16* qrowp = Qs + (rtile * 16 + ln) * EMBD;
    __bf16* pw = Ps + w * (16 * NK);

    for (int kt = 0; kt < T_CTX / NK; ++kt) {
        __syncthreads();   // both buffers' consumers finished previous step
        if (w == 0) {
            if (kt + 1 < T_CTX / NK) {
                const int t1 = (kt + 1) * NK;
                const unsigned bi = (unsigned)((kt + 1) & 1);
                tdm_load_2d(Kp + bb + (size_t)t1 * EMBD, kb_off + bi * (NK * EMBD * 2),
                            EMBD, T_CTX, EMBD, NK, EMBD);
                tdm_load_2d(Vtp + bb + t1, vb_off + bi * (NK * EMBD * 2),
                            T_CTX, EMBD, NK, EMBD, T_CTX);
                __builtin_amdgcn_s_wait_tensorcnt(2);  // tile kt (and Q) complete
            } else {
                __builtin_amdgcn_s_wait_tensorcnt(0);
            }
        }
        __syncthreads();

        const __bf16* Ks = Kb + (kt & 1) * (NK * EMBD);
        const __bf16* Vt = Vb + (kt & 1) * (NK * EMBD);

        // ---- S = Q K^T : 16 rows x 32 tokens, K-dim 512 = 16 wmma steps ----
        v8f s0 = zero, s1 = zero;
        #pragma unroll
        for (int ks = 0; ks < 16; ++ks) {
            const int kbm = ks * 32;
            v16bf a = ld_frag(qrowp + kbm + g * 16, qrowp + kbm + g * 16 + 8);
            const __bf16* k0 = Ks + ln * EMBD + kbm + g * 16;
            const __bf16* k1 = Ks + (16 + ln) * EMBD + kbm + g * 16;
            v16bf b0 = ld_frag(k0, k0 + 8);
            v16bf b1 = ld_frag(k1, k1 + 8);
            s0 = __builtin_amdgcn_wmma_f32_16x16x32_bf16(false, a, false, b0,
                                                         (short)0, s0, false, false);
            s1 = __builtin_amdgcn_wmma_f32_16x16x32_bf16(false, a, false, b1,
                                                         (short)0, s1, false, false);
        }

        // ---- online softmax ----
        #pragma unroll
        for (int r = 0; r < 8; ++r) {
            float t  = redmax16(fmaxf(s0[r], s1[r])) * C2;
            float mn = fmaxf(mrow[r], t);
            float alpha = exp2f(mrow[r] - mn);
            float p0 = exp2f(s0[r] * C2 - mn);
            float p1 = exp2f(s1[r] * C2 - mn);
            float rs = redsum16(p0 + p1);
            lrow[r] = lrow[r] * alpha + rs;
            mrow[r] = mn;
            const int prow = r + 8 * g;
            pw[prow * NK + ln]      = (__bf16)p0;
            pw[prow * NK + 16 + ln] = (__bf16)p1;
            #pragma unroll
            for (int nt = 0; nt < 16; ++nt) acc[nt][r] *= alpha;
        }

        // ---- O += P V ----
        v16bf ap = ld_frag(pw + ln * NK + g * 16, pw + ln * NK + g * 16 + 8);
        #pragma unroll
        for (int nt = 0; nt < 16; ++nt) {
            const __bf16* vp = Vt + (dbase + nt * 16 + ln) * NK + g * 16;
            v16bf bv = ld_frag(vp, vp + 8);
            acc[nt] = __builtin_amdgcn_wmma_f32_16x16x32_bf16(false, ap, false, bv,
                                                              (short)0, acc[nt], false, false);
        }
    }

    // ---- normalize and store with inverse (same) permutation ----
    #pragma unroll
    for (int r = 0; r < 8; ++r) {
        const float inv = 1.0f / lrow[r];
        const int qrow = m0 + rtile * 16 + r + 8 * g;
        float* op = Out + bb + (size_t)sperm(qrow) * EMBD + dbase + ln;
        #pragma unroll
        for (int nt = 0; nt < 16; ++nt) op[nt * 16] = acc[nt][r] * inv;
    }
}

extern "C" void kernel_launch(void* const* d_in, const int* in_sizes, int n_in,
                              void* d_out, int out_size, void* d_ws, size_t ws_size,
                              hipStream_t stream) {
    (void)in_sizes; (void)n_in; (void)out_size; (void)ws_size;
    const float* q = (const float*)d_in[0];
    const float* k = (const float*)d_in[1];
    const float* v = (const float*)d_in[2];
    float* out = (float*)d_out;

    // workspace: Qp 16MB | Kp 16MB | Vtp 16MB (bf16)
    const size_t elems = (size_t)4 * T_CTX * EMBD;
    unsigned short* Qp  = (unsigned short*)d_ws;
    unsigned short* Kp  = Qp + elems;
    unsigned short* Vtp = Kp + elems;

    dim3 block(256);
    dim3 grid(4 * (T_CTX / MB));   // 256 workgroups

    prep_kernel<<<grid, block, 0, stream>>>(q, k, v, Qp, Kp, Vtp);

    // Qs 64K + K dbl 64K + V^T dbl 64K + P 8K = 200 KB dynamic LDS
    size_t shmem = (size_t)(MB * EMBD + 2 * NK * EMBD + 2 * EMBD * NK + 8 * 16 * NK) * 2;
    strided_attn_fa<<<grid, block, shmem, stream>>>(Qp, Kp, Vtp, out);
}
